// SparseDispatcher_20229295964740
// MI455X (gfx1250) — matrix operand order repaired and checked by reading, not compiled, not run on hardware
//
#include <hip/hip_runtime.h>
#include <hip/hip_bf16.h>

// ---------------------------------------------------------------- constants
#define T_    8192      // B*S tokens
#define D_    1024
#define H_    4096
#define O_    1024
#define E_    8
#define KTOP_ 2
#define CAP_  1536
#define LNEPS 1e-5f

// CDNA5 async global->LDS staging (global_load_async_to_lds_b128 + ASYNCcnt).
#define USE_ASYNC_COPY 1

typedef __attribute__((ext_vector_type(16))) __bf16 v16bf;
typedef __attribute__((ext_vector_type(8)))  float  v8f;

union FragAB { v16bf v; uint4 q[2]; };

__device__ __forceinline__ unsigned short f2bf(float f) {
    unsigned int u = __float_as_uint(f);
    unsigned int r = u + 0x7FFFu + ((u >> 16) & 1u);
    return (unsigned short)(r >> 16);
}

__device__ __forceinline__ float gelu_exact(float v) {
    return 0.5f * v * (1.0f + erff(v * 0.70710678118654752f));
}

// ------------------------------------------------------- fp32 -> bf16 convert
__global__ void cvt_f32_bf16(const float* __restrict__ in,
                             unsigned short* __restrict__ out, size_t n) {
    size_t i = (size_t)blockIdx.x * blockDim.x + threadIdx.x;
    size_t stride = (size_t)gridDim.x * blockDim.x;
    for (; i < n; i += stride) out[i] = f2bf(in[i]);
}

// -------------------------------------- weight pack: fp32 [E,K,N] row-major
// -> bf16 WMMA-B-fragment-native layout [E][K/32][N][32]  (32 = K-panel)
__global__ __launch_bounds__(256)
void pack_w_bf16(const float* __restrict__ in, unsigned short* __restrict__ out,
                 int K, int N) {
    const int e  = blockIdx.z;
    const int k0 = blockIdx.y * 32;
    const int n0 = blockIdx.x * 32;
    const int tid = threadIdx.x;
    __shared__ unsigned short tile[32 * 33];
    const float* src = in + ((size_t)e * K + k0) * N + n0;
#pragma unroll
    for (int j = 0; j < 4; ++j) {
        int idx = tid + j * 256;            // 0..1023
        int kk = idx >> 5, nn = idx & 31;
        tile[kk * 33 + nn] = f2bf(src[(size_t)kk * N + nn]);
    }
    __syncthreads();
    unsigned short* dst =
        out + (((size_t)e * (K / 32) + (k0 >> 5)) * (size_t)N + n0) * 32;
#pragma unroll
    for (int j = 0; j < 4; ++j) {
        int idx = tid + j * 256;            // linear over [n_local][k_local]
        int nl = idx >> 5, kl = idx & 31;
        dst[idx] = tile[kl * 33 + nl];
    }
}

// ------------------------------------------------------------------- router
__global__ __launch_bounds__(256)
void router_kernel(const float* __restrict__ x,
                   const float* __restrict__ rw,
                   const float* __restrict__ rb,
                   int*   __restrict__ slot_e,   // [T*KTOP]
                   float* __restrict__ slot_w) { // [T*KTOP]
    const int wv = threadIdx.x >> 5;
    const int lane = threadIdx.x & 31;
    const int t = blockIdx.x * 8 + wv;
    const float* xr = x + (size_t)t * D_;

    float a[E_];
#pragma unroll
    for (int e = 0; e < E_; ++e) a[e] = 0.0f;
    for (int d = lane; d < D_; d += 32) {
        float xv = xr[d];
#pragma unroll
        for (int e = 0; e < E_; ++e) a[e] = fmaf(xv, rw[e * D_ + d], a[e]);
    }
#pragma unroll
    for (int e = 0; e < E_; ++e) {
        float v = a[e];
#pragma unroll
        for (int off = 16; off > 0; off >>= 1) v += __shfl_xor(v, off, 32);
        a[e] = v;
    }
    if (lane == 0) {
        float m = -3.402823e38f;
#pragma unroll
        for (int e = 0; e < E_; ++e) { a[e] += rb[e]; m = fmaxf(m, a[e]); }
        float p[E_];
#pragma unroll
        for (int e = 0; e < E_; ++e) p[e] = __expf(a[e] - m);
        int i1 = 0;
#pragma unroll
        for (int e = 1; e < E_; ++e) if (p[e] > p[i1]) i1 = e;
        int i2 = (i1 == 0) ? 1 : 0;
#pragma unroll
        for (int e = 0; e < E_; ++e) if (e != i1 && p[e] > p[i2]) i2 = e;
        float s = p[i1] + p[i2];
        slot_e[t * 2 + 0] = i1; slot_w[t * 2 + 0] = p[i1] / s;
        slot_e[t * 2 + 1] = i2; slot_w[t * 2 + 1] = p[i2] / s;
    }
}

// -------------------------------------------- deterministic capacity select
__global__ __launch_bounds__(256)
void select_kernel(const int* __restrict__ slot_e,
                   const float* __restrict__ slot_w,
                   int*   __restrict__ tok,    // [E, CAP]
                   float* __restrict__ wgt) {  // [E, CAP]
    const int e = blockIdx.x;
    const int tid = threadIdx.x;
    const int lane = tid & 31, wv = tid >> 5;
    __shared__ int wsum[8];
    int running = 0;
    const int NS = T_ * KTOP_;
    for (int base = 0; base < NS; base += 256) {
        int s = base + tid;
        bool match = (slot_e[s] == e);
        unsigned long long bal = __ballot(match);
        int inwave = __popcll(bal & ((1ull << lane) - 1ull));
        if (lane == 0) wsum[wv] = __popcll(bal);
        __syncthreads();
        int wpre = 0, total = 0;
#pragma unroll
        for (int j = 0; j < 8; ++j) { int c = wsum[j]; total += c; if (j < wv) wpre += c; }
        int idx = running + wpre + inwave;
        if (match && idx < CAP_) {
            tok[e * CAP_ + idx] = s >> 1;
            wgt[e * CAP_ + idx] = slot_w[s];
        }
        running += total;
        __syncthreads();
    }
}

// -------------------------------------------------------- A-tile staging
// Each thread copies 2x16B of one gathered row into LDS. The LDS address
// operand is derived from the real array pointer (low 32 bits of the generic
// address == group-segment offset), so the array escapes into the asm and the
// compiler must honor the asm's may-write of LDS (keeps the ds_load reads).
__device__ __forceinline__ void stage_a(const unsigned short* gsrc,
                                        unsigned short* lds_dst) {
#if USE_ASYNC_COPY
    unsigned int laddr = (unsigned int)(unsigned long long)lds_dst;
    asm volatile("global_load_async_to_lds_b128 %0, %1, off"
                 :: "v"(laddr), "v"(gsrc) : "memory");
    asm volatile("global_load_async_to_lds_b128 %0, %1, off offset:16"
                 :: "v"(laddr + 16), "v"(gsrc) : "memory");
#else
    uint4 a0 = *(const uint4*)(gsrc);
    uint4 a1 = *(const uint4*)(gsrc + 8);
    *(uint4*)(lds_dst)     = a0;
    *(uint4*)(lds_dst + 8) = a1;
#endif
}

__device__ __forceinline__ void stage_a_wait() {
#if USE_ASYNC_COPY
    asm volatile("s_wait_asynccnt 0x0" ::: "memory");
#endif
}

// ------------------------------------------------- GEMM1: h = gelu(xe@w1+b1)
// Block: 256 thr (8 waves), 128x64 tile; wave owns 32x32 (4 WMMAs / K-step).
__global__ __launch_bounds__(256)
void moe_gemm1(const unsigned short* __restrict__ xb,   // [T, D] bf16
               const unsigned short* __restrict__ w1p,  // [E][D/32][H][32] bf16
               const float* __restrict__ b1,            // [E, H]
               const int* __restrict__ tok,             // [E, CAP]
               unsigned short* __restrict__ hbuf) {     // [E, CAP, H] bf16
    const int e  = blockIdx.z;
    const int n0 = blockIdx.x * 64;
    const int m0 = blockIdx.y * 128;
    __shared__ unsigned short ldsA[128 * 32];           // [m][k], 8KB
    const int tid = threadIdx.x;
    const int lane = tid & 31, wv = tid >> 5;

    const int arow  = tid >> 1;                         // 0..127
    const int achnk = (tid & 1) * 16;                   // 0 or 16
    const int mytok = tok[e * CAP_ + m0 + arow];
    const unsigned short* aptr = xb + (size_t)mytok * D_ + achnk;
    unsigned short* a_lds_dst = &ldsA[arow * 32 + achnk];

    const int mt0 = (wv >> 1) * 32;                     // 0,32,64,96
    const int nt0 = (wv & 1) * 32;                      // 0,32
    const int mlane = lane & 15;
    const int kba = (lane >> 4) * 16;
    const int kbb = (lane >> 4) * 16;
    const int nglob0 = n0 + nt0 + mlane;
    const int nglob1 = nglob0 + 16;

    const unsigned short* bpanel = w1p + (size_t)e * (D_ / 32) * H_ * 32;

    v8f acc00 = {}; v8f acc01 = {}; v8f acc10 = {}; v8f acc11 = {};

    for (int k0 = 0; k0 < D_; k0 += 32) {
        stage_a(aptr + k0, a_lds_dst);
        const unsigned short* b0p =
            bpanel + (((size_t)(k0 >> 5) * H_ + nglob0) << 5) + kbb;
        const unsigned short* b1p =
            bpanel + (((size_t)(k0 >> 5) * H_ + nglob1) << 5) + kbb;
        FragAB fb0, fb1;
        fb0.q[0] = *(const uint4*)(b0p);
        fb0.q[1] = *(const uint4*)(b0p + 8);
        fb1.q[0] = *(const uint4*)(b1p);
        fb1.q[1] = *(const uint4*)(b1p + 8);
        if (k0 + 32 < D_) {
            __builtin_prefetch(aptr + k0 + 32, 0, 1);
            __builtin_prefetch(b0p + (size_t)H_ * 32, 0, 1);
            __builtin_prefetch(b1p + (size_t)H_ * 32, 0, 1);
        }
        stage_a_wait();
        __syncthreads();

        FragAB fa0, fa1;
        fa0.q[0] = *(const uint4*)(&ldsA[(mt0 + mlane) * 32 + kba]);
        fa0.q[1] = *(const uint4*)(&ldsA[(mt0 + mlane) * 32 + kba + 8]);
        fa1.q[0] = *(const uint4*)(&ldsA[(mt0 + 16 + mlane) * 32 + kba]);
        fa1.q[1] = *(const uint4*)(&ldsA[(mt0 + 16 + mlane) * 32 + kba + 8]);

        acc00 = __builtin_amdgcn_wmma_f32_16x16x32_bf16(false, fa0.v, false, fb0.v,
                                                        (short)0, acc00, false, false);
        acc01 = __builtin_amdgcn_wmma_f32_16x16x32_bf16(false, fa0.v, false, fb1.v,
                                                        (short)0, acc01, false, false);
        acc10 = __builtin_amdgcn_wmma_f32_16x16x32_bf16(false, fa1.v, false, fb0.v,
                                                        (short)0, acc10, false, false);
        acc11 = __builtin_amdgcn_wmma_f32_16x16x32_bf16(false, fa1.v, false, fb1.v,
                                                        (short)0, acc11, false, false);
        __syncthreads();
    }

    const float bias0 = b1[e * H_ + nglob0];
    const float bias1 = b1[e * H_ + nglob1];
    const int rbase = (lane >> 4) * 8;
#pragma unroll
    for (int r = 0; r < 8; ++r) {
        int mr0 = m0 + mt0 + rbase + r;
        int mr1 = mr0 + 16;
        size_t row0 = ((size_t)e * CAP_ + mr0) * H_;
        size_t row1 = ((size_t)e * CAP_ + mr1) * H_;
        hbuf[row0 + nglob0] = f2bf(gelu_exact(acc00[r] + bias0));
        hbuf[row0 + nglob1] = f2bf(gelu_exact(acc01[r] + bias1));
        hbuf[row1 + nglob0] = f2bf(gelu_exact(acc10[r] + bias0));
        hbuf[row1 + nglob1] = f2bf(gelu_exact(acc11[r] + bias1));
    }
}

// --------------------------------- GEMM2: ye = (h@w2+b2)*wgt, scatter-add
__global__ __launch_bounds__(256)
void moe_gemm2(const unsigned short* __restrict__ hbuf, // [E, CAP, H] bf16
               const unsigned short* __restrict__ w2p,  // [E][H/32][O][32] bf16
               const float* __restrict__ b2,            // [E, O]
               const int* __restrict__ tok,             // [E, CAP]
               const float* __restrict__ wgt,           // [E, CAP]
               float* __restrict__ accb) {              // [T, O] fp32
    const int e  = blockIdx.z;
    const int n0 = blockIdx.x * 64;
    const int m0 = blockIdx.y * 128;
    __shared__ unsigned short ldsA[128 * 32];
    const int tid = threadIdx.x;
    const int lane = tid & 31, wv = tid >> 5;

    const int arow  = tid >> 1;
    const int achnk = (tid & 1) * 16;
    const unsigned short* aptr =
        hbuf + ((size_t)e * CAP_ + m0 + arow) * H_ + achnk;
    unsigned short* a_lds_dst = &ldsA[arow * 32 + achnk];

    const int mt0 = (wv >> 1) * 32;
    const int nt0 = (wv & 1) * 32;
    const int mlane = lane & 15;
    const int kba = (lane >> 4) * 16;
    const int kbb = (lane >> 4) * 16;
    const int nglob0 = n0 + nt0 + mlane;
    const int nglob1 = nglob0 + 16;

    const unsigned short* bpanel = w2p + (size_t)e * (H_ / 32) * O_ * 32;

    v8f acc00 = {}; v8f acc01 = {}; v8f acc10 = {}; v8f acc11 = {};

    for (int k0 = 0; k0 < H_; k0 += 32) {
        stage_a(aptr + k0, a_lds_dst);
        const unsigned short* b0p =
            bpanel + (((size_t)(k0 >> 5) * O_ + nglob0) << 5) + kbb;
        const unsigned short* b1p =
            bpanel + (((size_t)(k0 >> 5) * O_ + nglob1) << 5) + kbb;
        FragAB fb0, fb1;
        fb0.q[0] = *(const uint4*)(b0p);
        fb0.q[1] = *(const uint4*)(b0p + 8);
        fb1.q[0] = *(const uint4*)(b1p);
        fb1.q[1] = *(const uint4*)(b1p + 8);
        if (k0 + 32 < H_) {
            __builtin_prefetch(aptr + k0 + 32, 0, 1);
            __builtin_prefetch(b0p + (size_t)O_ * 32, 0, 1);
            __builtin_prefetch(b1p + (size_t)O_ * 32, 0, 1);
        }
        stage_a_wait();
        __syncthreads();

        FragAB fa0, fa1;
        fa0.q[0] = *(const uint4*)(&ldsA[(mt0 + mlane) * 32 + kba]);
        fa0.q[1] = *(const uint4*)(&ldsA[(mt0 + mlane) * 32 + kba + 8]);
        fa1.q[0] = *(const uint4*)(&ldsA[(mt0 + 16 + mlane) * 32 + kba]);
        fa1.q[1] = *(const uint4*)(&ldsA[(mt0 + 16 + mlane) * 32 + kba + 8]);

        acc00 = __builtin_amdgcn_wmma_f32_16x16x32_bf16(false, fa0.v, false, fb0.v,
                                                        (short)0, acc00, false, false);
        acc01 = __builtin_amdgcn_wmma_f32_16x16x32_bf16(false, fa0.v, false, fb1.v,
                                                        (short)0, acc01, false, false);
        acc10 = __builtin_amdgcn_wmma_f32_16x16x32_bf16(false, fa1.v, false, fb0.v,
                                                        (short)0, acc10, false, false);
        acc11 = __builtin_amdgcn_wmma_f32_16x16x32_bf16(false, fa1.v, false, fb1.v,
                                                        (short)0, acc11, false, false);
        __syncthreads();
    }

    const float bias0 = b2[e * O_ + nglob0];
    const float bias1 = b2[e * O_ + nglob1];
    const int rbase = (lane >> 4) * 8;
#pragma unroll
    for (int r = 0; r < 8; ++r) {
        int mr0 = m0 + mt0 + rbase + r;
        int mr1 = mr0 + 16;
        float w0 = wgt[e * CAP_ + mr0];
        float w1 = wgt[e * CAP_ + mr1];
        if (w0 > 0.0f) {
            size_t t0 = (size_t)tok[e * CAP_ + mr0] * O_;
            atomicAdd(&accb[t0 + nglob0], (acc00[r] + bias0) * w0);
            atomicAdd(&accb[t0 + nglob1], (acc01[r] + bias1) * w0);
        }
        if (w1 > 0.0f) {
            size_t t1 = (size_t)tok[e * CAP_ + mr1] * O_;
            atomicAdd(&accb[t1 + nglob0], (acc10[r] + bias0) * w1);
            atomicAdd(&accb[t1 + nglob1], (acc11[r] + bias1) * w1);
        }
    }
}

// ---------------------------------------------------------------- LayerNorm
__global__ __launch_bounds__(256)
void ln_kernel(const float* __restrict__ accb,
               const float* __restrict__ g, const float* __restrict__ b,
               float* __restrict__ out) {
    const int t = blockIdx.x;
    const float* row = accb + (size_t)t * O_;
    const int tid = threadIdx.x;
    const int lane = tid & 31, wv = tid >> 5;
    __shared__ float sm[8], sm2[8];
    float s = 0.f, ss = 0.f;
    for (int i = tid; i < O_; i += 256) {
        float v = row[i];
        s += v; ss += v * v;
    }
#pragma unroll
    for (int off = 16; off > 0; off >>= 1) {
        s  += __shfl_xor(s, off, 32);
        ss += __shfl_xor(ss, off, 32);
    }
    if (lane == 0) { sm[wv] = s; sm2[wv] = ss; }
    __syncthreads();
    float ts = 0.f, tss = 0.f;
#pragma unroll
    for (int j = 0; j < 8; ++j) { ts += sm[j]; tss += sm2[j]; }
    float mu = ts * (1.0f / O_);
    float var = tss * (1.0f / O_) - mu * mu;
    float rstd = rsqrtf(var + LNEPS);
    for (int i = tid; i < O_; i += 256) {
        out[(size_t)t * O_ + i] = (row[i] - mu) * rstd * g[i] + b[i];
    }
}

// ------------------------------------------------------------------ launcher
extern "C" void kernel_launch(void* const* d_in, const int* in_sizes, int n_in,
                              void* d_out, int out_size, void* d_ws, size_t ws_size,
                              hipStream_t stream) {
    const float* x        = (const float*)d_in[0];
    const float* router_w = (const float*)d_in[1];
    const float* router_b = (const float*)d_in[2];
    const float* w1       = (const float*)d_in[3];
    const float* b1       = (const float*)d_in[4];
    const float* w2       = (const float*)d_in[5];
    const float* b2       = (const float*)d_in[6];
    const float* ln_g     = (const float*)d_in[7];
    const float* ln_b     = (const float*)d_in[8];
    float* out = (float*)d_out;

    char* ws = (char*)d_ws;
    size_t off = 0;
    auto take = [&](size_t bytes) {
        void* p = ws + off;
        off += (bytes + 255) & ~(size_t)255;
        return p;
    };
    unsigned short* xb   = (unsigned short*)take((size_t)T_ * D_ * 2);
    unsigned short* w1p  = (unsigned short*)take((size_t)E_ * D_ * H_ * 2);
    unsigned short* w2p  = (unsigned short*)take((size_t)E_ * H_ * O_ * 2);
    unsigned short* hbuf = (unsigned short*)take((size_t)E_ * CAP_ * H_ * 2);
    float* accb          = (float*)take((size_t)T_ * O_ * 4);
    int*   slot_e        = (int*)take((size_t)T_ * KTOP_ * 4);
    float* slot_w        = (float*)take((size_t)T_ * KTOP_ * 4);
    int*   tok           = (int*)take((size_t)E_ * CAP_ * 4);
    float* wgt           = (float*)take((size_t)E_ * CAP_ * 4);

    hipMemsetAsync(accb, 0, (size_t)T_ * O_ * 4, stream);
    hipMemsetAsync(tok,  0, (size_t)E_ * CAP_ * 4, stream);
    hipMemsetAsync(wgt,  0, (size_t)E_ * CAP_ * 4, stream);

    cvt_f32_bf16<<<4096, 256, 0, stream>>>(x, xb, (size_t)T_ * D_);
    pack_w_bf16<<<dim3(H_ / 32, D_ / 32, E_), 256, 0, stream>>>(w1, w1p, D_, H_);
    pack_w_bf16<<<dim3(O_ / 32, H_ / 32, E_), 256, 0, stream>>>(w2, w2p, H_, O_);

    router_kernel<<<T_ / 8, 256, 0, stream>>>(x, router_w, router_b, slot_e, slot_w);
    select_kernel<<<E_, 256, 0, stream>>>(slot_e, slot_w, tok, wgt);

    moe_gemm1<<<dim3(H_ / 64, CAP_ / 128, E_), 256, 0, stream>>>(xb, w1p, b1, tok, hbuf);
    moe_gemm2<<<dim3(O_ / 64, CAP_ / 128, E_), 256, 0, stream>>>(hbuf, w2p, b2, tok, wgt, accb);

    ln_kernel<<<T_, 256, 0, stream>>>(accb, ln_g, ln_b, out);
}
